// SCNN_31413390803564
// MI455X (gfx1250) — matrix operand order repaired and checked by reading, not compiled, not run on hardware
//
#include <hip/hip_runtime.h>
#include <hip/hip_bf16.h>

// ---------------------------------------------------------------------------
// SCNN message passing on MI455X (gfx1250, wave32, WMMA).
//
// x: [B=32, C=128, H=36, W=100] f32. Four sequential directional scans; each
// step is a 128->128 9-tap conv = 9 accumulated K=128 GEMMs done with
// V_WMMA_F32_16X16X32_BF16 (bf16 in, f32 accumulate).
//
// Structure per step (one kernel launch per recurrence step, graph-captured):
//   - one block per batch image b (32 blocks, 256 thr = 8 waves = 8 M-tiles)
//   - whole source line + halo staged to LDS as bf16, transposed [col][ci]
//     with 272 B column stride so every B fragment is one 16B-aligned 32 B
//     contiguous LDS read (2x ds_load_b128)
//   - per (tap k, K-chunk q): load the A fragment ONCE, sweep it over all
//     N-tiles with independent accumulators -> 7x/3x less weight traffic and
//     deep independent WMMA chains for XDL pipelining.
//
// Whole tensor (59 MB) + packed weights (1.1 MB) stay resident in the 192 MB
// L2 across all 266 steps; HBM is touched only at the ends.
// ---------------------------------------------------------------------------

typedef __attribute__((ext_vector_type(16))) __bf16 v16bf;
typedef __attribute__((ext_vector_type(8)))  __bf16 v8bf;
typedef __attribute__((ext_vector_type(8)))  float  v8f;

#define B_   32
#define C_   128
#define H_   36
#define W_   100
#define KW   9
#define PAD  4

// LDS column stride in bf16 elements: 136 -> 272 bytes, multiple of 16 so
// (col*136 + q*32 + kb)*2 is 16B-aligned for ds_load_b128 fragment reads.
#define LDS_STRIDE 136

#define NLT_ROW 7   // ceil(100/16) N-tiles along W
#define NLT_COL 3   // ceil(36/16)  N-tiles along H
#define COLS_ROW 120 // staged columns: l in [-4, 115] (>=100 zero-padded)
#define COLS_COL 56  // staged columns: h in [-4, 51]  (>=36 zero-padded)

// Packed A-fragment layout: [dir][mt(8)][tap(9)][q(4)][lane(32)][16 x bf16]
__device__ __forceinline__ size_t wpack_idx(int dir, int mt, int k, int q, int lane) {
    return ((((size_t)((dir * 8 + mt) * 9 + k)) * 4 + q) * 32 + lane) * 16;
}

// ---------------------------------------------------------------------------
// Pack fp32 weights [Co=128][Ci=128][9] into WMMA 16x32 bf16 A-fragments.
// 16-bit A 16x32, contiguous K order (matches B): lane L holds M = L%16;
//   VGPR i -> K = 2i,2i+1 ; L>=16: K += 16
// ---------------------------------------------------------------------------
__global__ void pack_weights(const float* __restrict__ wd,
                             const float* __restrict__ wu,
                             const float* __restrict__ wr,
                             const float* __restrict__ wl,
                             __bf16* __restrict__ packed) {
    int t = blockIdx.x * blockDim.x + threadIdx.x;
    const int total = 4 * 8 * 9 * 4 * 32 * 8;   // dirs*mt*tap*q*lane*vgpr
    if (t >= total) return;
    int i2   = t & 7;           // vgpr index (bf16 pair)
    int L    = (t >> 3) & 31;   // lane
    int q    = (t >> 8) & 3;    // 32-wide K chunk (ci block)
    int k    = (t >> 10) % 9;   // tap
    int rest = (t >> 10) / 9;
    int mt   = rest & 7;        // M tile (co block)
    int d    = rest >> 3;       // direction

    const float* w = (d == 0) ? wd : (d == 1) ? wu : (d == 2) ? wr : wl;

    int K0  = 2 * i2 + ((L < 16) ? 0 : 16);
    int co  = mt * 16 + (L & 15);
    int ci0 = q * 32 + K0;

    size_t base = wpack_idx(d, mt, k, q, L);
    packed[base + 2 * i2]     = (__bf16)w[(co * C_ + ci0) * KW + k];
    packed[base + 2 * i2 + 1] = (__bf16)w[(co * C_ + ci0 + 1) * KW + k];
}

// Load one B fragment: 32 contiguous bytes of LDS, 16B-aligned -> 2x b128.
__device__ __forceinline__ v16bf load_bfrag(const __bf16* p) {
    union { v16bf v; v8bf h[2]; } u;
    u.h[0] = *(const v8bf*)(p);
    u.h[1] = *(const v8bf*)(p + 8);
    return u.v;
}

// ---------------------------------------------------------------------------
// One step of the vertical (down/up) scan over rows:
//   x[b, :, ht, :] += relu(convW_9(x[b, :, hs, :]))
// grid = 32 (one block per b), block = 256.
// ---------------------------------------------------------------------------
__global__ __launch_bounds__(256)
void row_step(float* __restrict__ x, const __bf16* __restrict__ packed,
              int dir, int ht, int hs) {
    __shared__ alignas(16) __bf16 lds[COLS_ROW * LDS_STRIDE];
    const int tid = threadIdx.x;
    const int b   = blockIdx.x;

    // Stage source row hs: cols 0..119 <-> l = -4..115, zero outside [0,W).
    const float* src = x + (((size_t)b * C_) * H_ + hs) * W_;
    for (int i = tid; i < COLS_ROW * C_; i += 256) {
        int ci  = i / COLS_ROW;
        int col = i - ci * COLS_ROW;
        int l   = col - PAD;
        float v = (l >= 0 && l < W_) ? src[(size_t)ci * (H_ * W_) + l] : 0.f;
        lds[col * LDS_STRIDE + ci] = (__bf16)v;
    }
    __syncthreads();

    const int lane = tid & 31;
    const int mt   = tid >> 5;
    const int n    = lane & 15;
    const int kb   = (lane < 16) ? 0 : 16;   // B-fragment K half

    v8f acc[NLT_ROW];
#pragma unroll
    for (int t = 0; t < NLT_ROW; ++t)
        acc[t] = (v8f){0.f, 0.f, 0.f, 0.f, 0.f, 0.f, 0.f, 0.f};

    for (int k = 0; k < KW; ++k) {
#pragma unroll
        for (int q = 0; q < 4; ++q) {
            v16bf a = *(const v16bf*)(packed + wpack_idx(dir, mt, k, q, lane));
            const __bf16* bbase = lds + (n + k) * LDS_STRIDE + q * 32 + kb;
#pragma unroll
            for (int t = 0; t < NLT_ROW; ++t) {
                v16bf bf = load_bfrag(bbase + t * (16 * LDS_STRIDE));
                acc[t] = __builtin_amdgcn_wmma_f32_16x16x32_bf16(
                    false, a, false, bf, (short)0, acc[t], false, false);
            }
        }
    }

    // Epilogue: relu + in-place accumulate into target row ht.
    float* dstb = x + (((size_t)b * C_ + mt * 16) * H_ + ht) * W_;
#pragma unroll
    for (int t = 0; t < NLT_ROW; ++t) {
        int l = t * 16 + n;
        if (l < W_) {
#pragma unroll
            for (int r = 0; r < 8; ++r) {
                int   m = r + ((lane < 16) ? 0 : 8);
                float f = acc[t][r];
                f = f > 0.f ? f : 0.f;
                float* p = dstb + (size_t)m * (H_ * W_) + l;
                *p = *p + f;
            }
        }
    }
}

// ---------------------------------------------------------------------------
// One step of the horizontal (right/left) scan over columns:
//   x[b, :, :, wt] += relu(convH_9(x[b, :, :, ws]))
// grid = 32 (one block per b), block = 256.
// ---------------------------------------------------------------------------
__global__ __launch_bounds__(256)
void col_step(float* __restrict__ x, const __bf16* __restrict__ packed,
              int dir, int wt, int ws) {
    __shared__ alignas(16) __bf16 lds[COLS_COL * LDS_STRIDE];
    const int tid = threadIdx.x;
    const int b   = blockIdx.x;

    // Stage source column ws: cols 0..55 <-> h = -4..51, zero outside [0,H).
    const float* src = x + (((size_t)b * C_) * H_) * W_ + ws;
    for (int i = tid; i < COLS_COL * C_; i += 256) {
        int ci  = i / COLS_COL;
        int col = i - ci * COLS_COL;
        int h   = col - PAD;
        float v = (h >= 0 && h < H_) ? src[((size_t)ci * H_ + h) * W_] : 0.f;
        lds[col * LDS_STRIDE + ci] = (__bf16)v;
    }
    __syncthreads();

    const int lane = tid & 31;
    const int mt   = tid >> 5;
    const int n    = lane & 15;
    const int kb   = (lane < 16) ? 0 : 16;

    v8f acc[NLT_COL];
#pragma unroll
    for (int t = 0; t < NLT_COL; ++t)
        acc[t] = (v8f){0.f, 0.f, 0.f, 0.f, 0.f, 0.f, 0.f, 0.f};

    for (int k = 0; k < KW; ++k) {
#pragma unroll
        for (int q = 0; q < 4; ++q) {
            v16bf a = *(const v16bf*)(packed + wpack_idx(dir, mt, k, q, lane));
            const __bf16* bbase = lds + (n + k) * LDS_STRIDE + q * 32 + kb;
#pragma unroll
            for (int t = 0; t < NLT_COL; ++t) {
                v16bf bf = load_bfrag(bbase + t * (16 * LDS_STRIDE));
                acc[t] = __builtin_amdgcn_wmma_f32_16x16x32_bf16(
                    false, a, false, bf, (short)0, acc[t], false, false);
            }
        }
    }

    float* dstb = x + (((size_t)b * C_ + mt * 16) * H_) * W_ + wt;
#pragma unroll
    for (int t = 0; t < NLT_COL; ++t) {
        int h = t * 16 + n;
        if (h < H_) {
#pragma unroll
            for (int r = 0; r < 8; ++r) {
                int   m = r + ((lane < 16) ? 0 : 8);
                float f = acc[t][r];
                f = f > 0.f ? f : 0.f;
                float* p = dstb + ((size_t)m * H_ + h) * W_;
                *p = *p + f;
            }
        }
    }
}

// ---------------------------------------------------------------------------
extern "C" void kernel_launch(void* const* d_in, const int* in_sizes, int n_in,
                              void* d_out, int out_size, void* d_ws, size_t ws_size,
                              hipStream_t stream) {
    (void)in_sizes; (void)n_in; (void)out_size; (void)ws_size;
    const float* x_in = (const float*)d_in[0];
    const float* w_d  = (const float*)d_in[1];
    const float* w_u  = (const float*)d_in[2];
    const float* w_r  = (const float*)d_in[3];
    const float* w_l  = (const float*)d_in[4];

    float*  x      = (float*)d_out;                 // work in place on output
    __bf16* packed = (__bf16*)d_ws;                 // ~1.13 MB packed weights

    const size_t nbytes = sizeof(float) * (size_t)B_ * C_ * H_ * W_;
    hipMemcpyAsync(x, x_in, nbytes, hipMemcpyDeviceToDevice, stream);

    const int pack_total = 4 * 8 * 9 * 4 * 32 * 8;
    pack_weights<<<(pack_total + 255) / 256, 256, 0, stream>>>(w_d, w_u, w_r, w_l, packed);

    // down: rows 1..H-1, source i-1, weights w_d (dir 0)
    for (int i = 1; i < H_; ++i)
        row_step<<<B_, 256, 0, stream>>>(x, packed, 0, i, i - 1);
    // up: rows H-2..1 (first/last untouched), source i+1, weights w_u (dir 1)
    for (int i = H_ - 2; i >= 1; --i)
        row_step<<<B_, 256, 0, stream>>>(x, packed, 1, i, i + 1);
    // right: cols 1..W-1, source j-1, weights w_r (dir 2)
    for (int j = 1; j < W_; ++j)
        col_step<<<B_, 256, 0, stream>>>(x, packed, 2, j, j - 1);
    // left: cols W-2..1, source j+1, weights w_l (dir 3)
    for (int j = W_ - 2; j >= 1; --j)
        col_step<<<B_, 256, 0, stream>>>(x, packed, 3, j, j + 1);
}
